// Net_11553462026249
// MI455X (gfx1250) — matrix operand order repaired and checked by reading, not compiled, not run on hardware
//
#include <hip/hip_runtime.h>
#include <math.h>

// ---------------- problem constants ----------------
#define NB   64        // batch
#define TT   128       // sequence length
#define KK   256       // states
#define VV   32000     // vocab
#define EE   100       // embed dim
#define EP   128       // padded embed dim (bf16, multiple of 32 for WMMA K)
#define R    (KK*KK)   // 65536 rows of trans_w
#define CPB  8         // N-chunks (32 cols each) per block in step_gemm
#define LROW 136       // padded LDS row length in bf16 elems (272 B = 68 words)

typedef __attribute__((ext_vector_type(8)))  float    v8f;
typedef __attribute__((ext_vector_type(16))) __bf16   v16bf;
typedef __attribute__((ext_vector_type(4)))  unsigned u32x4;

union ABv { v16bf v; u32x4 q[2]; };

// float -> bf16 (round to nearest even)
__device__ __forceinline__ unsigned short f2bf(float f) {
    unsigned u = __builtin_bit_cast(unsigned, f);
    u += 0x7FFFu + ((u >> 16) & 1u);
    return (unsigned short)(u >> 16);
}

// async global->LDS copy of 16 bytes (CDNA5 ASYNCcnt path)
__device__ __forceinline__ void async_b128(unsigned ldsoff, const void* gaddr) {
    asm volatile("global_load_async_to_lds_b128 %0, %1, off"
                 :: "v"(ldsoff), "v"(gaddr) : "memory");
}

// ---------------- conversion / gather kernels ----------------
__global__ void conv_trans(const float* __restrict__ tw, unsigned short* __restrict__ Tb) {
    long idx = (long)blockIdx.x * 256 + threadIdx.x;          // R*EP elems
    if (idx >= (long)R * EP) return;
    int e = (int)(idx & (EP - 1));
    long r = idx >> 7;
    Tb[idx] = (e < EE) ? f2bf(tw[r * EE + e]) : (unsigned short)0;
}

__global__ void conv_x(const int* __restrict__ w, const float* __restrict__ emb,
                       unsigned short* __restrict__ Xb) {
    long idx = (long)blockIdx.x * 256 + threadIdx.x;          // (TT-1)*NB*EP elems
    if (idx >= (long)(TT - 1) * NB * EP) return;
    int e = (int)(idx & (EP - 1));
    int n = (int)((idx >> 7) & (NB - 1));
    int t = (int)(idx >> 13);                                  // 0..126 (x_prev time index)
    int wid = w[n * TT + t];
    Xb[idx] = (e < EE) ? f2bf(emb[(long)wid * EE + e]) : (unsigned short)0;
}

__global__ void conv_bf(const float* __restrict__ src, unsigned short* __restrict__ dst, long n) {
    long idx = (long)blockIdx.x * 256 + threadIdx.x;
    if (idx < n) dst[idx] = f2bf(src[idx]);
}

// ---------------- emission GEMM: score[k,v] = emb_cluster[k,:] . vocab_w[v,:] ----------------
__global__ void em_gemm(const unsigned short* __restrict__ Cb,
                        const unsigned short* __restrict__ Vb,
                        float* __restrict__ score) {
    const int lane = threadIdx.x & 31;
    const int wave = threadIdx.x >> 5;          // 0..7
    const int mt   = wave & 1;                  // 2 M-tiles per block
    const int nt   = wave >> 1;                 // 4 N-tiles per block
    const int mgrp = blockIdx.x / (VV / 64);    // 0..7
    const int vblk = blockIdx.x % (VV / 64);    // 0..499
    const int hi   = lane >> 4;
    const int l15  = lane & 15;

    const int mrow = mgrp * 32 + mt * 16 + l15;
    const int bcol = vblk * 64 + nt * 16 + l15;

    const unsigned short* arow = Cb + (long)mrow * KK;
    const unsigned short* brow = Vb + (long)bcol * KK;

    v8f c;
#pragma unroll
    for (int i = 0; i < 8; ++i) c[i] = 0.0f;

#pragma unroll
    for (int kb = 0; kb < KK; kb += 32) {
        __builtin_prefetch((const void*)(brow + kb + 32), 0, 1);  // global_prefetch_b8
        ABv a, b;
        a.q[0] = *(const u32x4*)(arow + kb + hi * 8);
        a.q[1] = *(const u32x4*)(arow + kb + hi * 8 + 16);
        b.q[0] = *(const u32x4*)(brow + kb + hi * 8);
        b.q[1] = *(const u32x4*)(brow + kb + hi * 8 + 16);
        c = __builtin_amdgcn_wmma_f32_16x16x32_bf16(false, a.v, false, b.v,
                                                    (short)0, c, false, false);
    }
    const int m0   = mgrp * 32 + mt * 16 + hi * 8;
    const int ncol = vblk * 64 + nt * 16 + l15;
#pragma unroll
    for (int v = 0; v < 8; ++v)
        score[(long)(m0 + v) * VV + ncol] = c[v];
}

// ---------------- Z[k] = logsumexp_v score[k,:] ----------------
__global__ void row_lse_V(const float* __restrict__ score, float* __restrict__ Z) {
    const int k = blockIdx.x;
    const float* r = score + (long)k * VV;
    float m = -INFINITY, s = 0.0f;
    for (int v = threadIdx.x; v < VV; v += 256) {
        float x = r[v];
        if (x > m) { s = s * __expf(m - x) + 1.0f; m = x; }
        else       { s += __expf(x - m); }
    }
    __shared__ float sm[256], ss[256];
    sm[threadIdx.x] = m; ss[threadIdx.x] = s;
    __syncthreads();
    for (int off = 128; off > 0; off >>= 1) {
        if ((int)threadIdx.x < off) {
            float m2 = sm[threadIdx.x + off], s2 = ss[threadIdx.x + off];
            float M  = fmaxf(sm[threadIdx.x], m2);
            ss[threadIdx.x] = ss[threadIdx.x] * __expf(sm[threadIdx.x] - M) + s2 * __expf(m2 - M);
            sm[threadIdx.x] = M;
        }
        __syncthreads();
    }
    if (threadIdx.x == 0) Z[k] = sm[0] + __logf(ss[0]);
}

// ---------------- gather per-token emissions: em_tok[t-1][n][k] ----------------
__global__ void em_gather(const float* __restrict__ score, const float* __restrict__ Z,
                          const int* __restrict__ w, float* __restrict__ em_tok) {
    long idx = (long)blockIdx.x * 256 + threadIdx.x;          // (TT-1)*NB*KK
    if (idx >= (long)(TT - 1) * NB * KK) return;
    int k   = (int)(idx & (KK - 1));
    int n   = (int)((idx >> 8) & (NB - 1));
    int tm1 = (int)(idx >> 14);
    int wid = w[n * TT + (tm1 + 1)];
    em_tok[idx] = score[(long)k * VV + wid] - Z[k];
}

// ---------------- pre_alpha = log_softmax(start_w + start_b), broadcast ----------------
__global__ void init_alpha(const float* __restrict__ sw, const float* __restrict__ sb,
                           float* __restrict__ alpha) {
    const int k = threadIdx.x;
    float x = sw[k] + sb[k];
    __shared__ float red[256];
    red[k] = x; __syncthreads();
    for (int off = 128; off > 0; off >>= 1) {
        if (k < off) red[k] = fmaxf(red[k], red[k + off]);
        __syncthreads();
    }
    float m = red[0]; __syncthreads();
    red[k] = __expf(x - m); __syncthreads();
    for (int off = 128; off > 0; off >>= 1) {
        if (k < off) red[k] += red[k + off];
        __syncthreads();
    }
    float val = x - (m + __logf(red[0]));
    for (int n = 0; n < NB; ++n) alpha[n * KK + k] = val;
}

// ---------------- per-step GEMM: S[n, r] = Xt[n,:] . Tb[r,:]  (bf16 WMMA) ----------------
// Pipelined: A panel staged once via async->LDS; B panels double-buffered with
// global_load_async_to_lds_b128 overlapped with WMMA on the previous chunk.
__global__ void step_gemm(const unsigned short* __restrict__ Xt,  // 64 x 128 bf16
                          const unsigned short* __restrict__ Tb,  // 65536 x 128 bf16
                          float* __restrict__ S) {                // 64 x 65536 f32
    __shared__ __align__(16) unsigned short sA[64 * LROW];        // 17.0 KB
    __shared__ __align__(16) unsigned short sB[2][32 * LROW];     // 17.0 KB

    const int tid  = threadIdx.x;
    const int lane = tid & 31;
    const int wave = tid >> 5;        // 0..7
    const int mt   = wave & 3;        // 4 M-tiles (all 64 batch rows)
    const int nt2  = wave >> 2;       // 2 N-tiles per chunk
    const int hi   = lane >> 4;
    const int l15  = lane & 15;

    const long colBase = (long)blockIdx.x * (32 * CPB);

    // stage A panel: 64 rows x 256 B = 1024 x 16B chunks (4 per thread)
#pragma unroll
    for (int q = tid; q < 1024; q += 256) {
        int row = q >> 4, cin = q & 15;
        async_b128((unsigned)(uintptr_t)&sA[row * LROW + cin * 8],
                   (const char*)Xt + row * 256 + cin * 16);
    }
    // stage B chunk 0: 32 rows x 256 B = 512 x 16B chunks (2 per thread)
#pragma unroll
    for (int q = tid; q < 512; q += 256) {
        int row = q >> 4, cin = q & 15;
        async_b128((unsigned)(uintptr_t)&sB[0][row * LROW + cin * 8],
                   (const char*)Tb + (colBase + row) * 256 + cin * 16);
    }

    for (int c = 0; c < CPB; ++c) {
        const int p = c & 1;
        if (c + 1 < CPB) {
            // prefetch next B chunk into the other buffer
#pragma unroll
            for (int q = tid; q < 512; q += 256) {
                int row = q >> 4, cin = q & 15;
                async_b128((unsigned)(uintptr_t)&sB[p ^ 1][row * LROW + cin * 8],
                           (const char*)Tb + (colBase + (long)(c + 1) * 32 + row) * 256 + cin * 16);
            }
            asm volatile("s_wait_asynccnt 2" ::: "memory");   // current chunk landed
        } else {
            asm volatile("s_wait_asynccnt 0" ::: "memory");
        }
        __syncthreads();

        const unsigned short* aL = &sA[(mt * 16 + l15) * LROW];
        const unsigned short* bL = &sB[p][(nt2 * 16 + l15) * LROW];

        v8f acc;
#pragma unroll
        for (int i = 0; i < 8; ++i) acc[i] = 0.0f;

#pragma unroll
        for (int kb = 0; kb < EP; kb += 32) {
            ABv a, b;
            a.q[0] = *(const u32x4*)(aL + kb + hi * 8);
            a.q[1] = *(const u32x4*)(aL + kb + hi * 8 + 16);
            b.q[0] = *(const u32x4*)(bL + kb + hi * 8);
            b.q[1] = *(const u32x4*)(bL + kb + hi * 8 + 16);
            acc = __builtin_amdgcn_wmma_f32_16x16x32_bf16(false, a.v, false, b.v,
                                                          (short)0, acc, false, false);
        }
        const int  m0   = mt * 16 + hi * 8;
        const long ncol = colBase + (long)c * 32 + nt2 * 16 + l15;
#pragma unroll
        for (int v = 0; v < 8; ++v)
            S[(long)(m0 + v) * R + ncol] = acc[v];
        __syncthreads();   // protect sB[p] before it is overwritten next iter
    }
}

// ---------------- per-step: rowl[n,i] = alpha[n,i] - lse_j S[n,i,:] ----------------
__global__ void row_logits(const float* __restrict__ S, const float* __restrict__ alpha,
                           float* __restrict__ rowl) {
    const int lane = threadIdx.x & 31;
    const int wave = threadIdx.x >> 5;
    const int row  = blockIdx.x * 8 + wave;   // 0..16383  (row = n*256 + i)
    const float* sr = S + (long)row * KK;

    float m = -INFINITY, s = 0.0f;
#pragma unroll
    for (int j = lane; j < KK; j += 32) {
        float x = sr[j];
        if (x > m) { s = s * __expf(m - x) + 1.0f; m = x; }
        else       { s += __expf(x - m); }
    }
#pragma unroll
    for (int off = 16; off > 0; off >>= 1) {
        float m2 = __shfl_xor(m, off, 32);
        float s2 = __shfl_xor(s, off, 32);
        float M  = fmaxf(m, m2);
        s = s * __expf(m - M) + s2 * __expf(m2 - M);
        m = M;
    }
    if (lane == 0) rowl[row] = alpha[row] - (m + __logf(s));
}

// ---------------- per-step: alpha[n,j] = lse_i(rowl[n,i] + S[n,i,j]) + em[n,j] ----------------
__global__ void col_combine(const float* __restrict__ S, const float* __restrict__ rowl,
                            const float* __restrict__ em_t, float* __restrict__ alpha) {
    const int n = blockIdx.x;
    const int j = threadIdx.x;
    const float* Sn = S + (long)n * R;
    __shared__ float srl[KK];
    srl[j] = rowl[n * KK + j];
    __syncthreads();

    float m = -INFINITY, s = 0.0f;
    for (int i = 0; i < KK; ++i) {
        float x = srl[i] + Sn[(long)i * KK + j];   // coalesced across j
        if (x > m) { s = s * __expf(m - x) + 1.0f; m = x; }
        else       { s += __expf(x - m); }
    }
    alpha[n * KK + j] = m + __logf(s) + em_t[n * KK + j];
}

// ---------------- final: -mean_n logsumexp_k alpha[n,:] ----------------
__global__ void final_reduce(const float* __restrict__ alpha, float* __restrict__ out) {
    __shared__ float sred[NB];
    const int t = threadIdx.x;
    if (t < NB) {
        const float* ar = alpha + t * KK;
        float m = -INFINITY, s = 0.0f;
        for (int k = 0; k < KK; ++k) {
            float x = ar[k];
            if (x > m) { s = s * __expf(m - x) + 1.0f; m = x; }
            else       { s += __expf(x - m); }
        }
        sred[t] = m + __logf(s);
    }
    __syncthreads();
    if (t == 0) {
        float acc = 0.0f;
        for (int n = 0; n < NB; ++n) acc += sred[n];
        out[0] = -acc / (float)NB;
    }
}

// ---------------- host orchestration ----------------
extern "C" void kernel_launch(void* const* d_in, const int* in_sizes, int n_in,
                              void* d_out, int out_size, void* d_ws, size_t ws_size,
                              hipStream_t stream) {
    const int*   w    = (const int*)  d_in[0];
    const float* emb  = (const float*)d_in[1];
    const float* voc  = (const float*)d_in[2];
    const float* ecw  = (const float*)d_in[3];
    const float* sw   = (const float*)d_in[4];
    const float* sb   = (const float*)d_in[5];
    const float* tw   = (const float*)d_in[6];
    float* out = (float*)d_out;
    char* ws = (char*)d_ws;

    // workspace layout (bytes, 256-aligned)
    const size_t TB_OFF    = 0;                                   // 65536*128*2  = 16,777,216
    const size_t XB_OFF    = TB_OFF + (size_t)R * EP * 2;         // 127*64*128*2 =  2,080,768
    const size_t AL_OFF    = XB_OFF + (size_t)(TT - 1) * NB * EP * 2;   // 64*256*4
    const size_t Z_OFF     = AL_OFF + (size_t)NB * KK * 4;        // 256*4 (pad to 1K)
    const size_t RL_OFF    = Z_OFF + 1024;                        // rowl 64*256*4
    const size_t EMT_OFF   = RL_OFF + (size_t)NB * KK * 4;        // 127*64*256*4 = 8,323,072
    const size_t UNION_OFF = EMT_OFF + (size_t)(TT - 1) * NB * KK * 4;
    // phase 1 (emission) view of the union region:
    const size_t CB_OFF    = UNION_OFF;                           // 256*256*2
    const size_t VB_OFF    = CB_OFF + (size_t)KK * KK * 2;        // 32000*256*2
    const size_t SC_OFF    = VB_OFF + (size_t)VV * KK * 2;        // 256*32000*4
    // phase 2 (scan) view: S aliases CB/VB and the first 256 KB of score (dead by then)
    const size_t S_OFF     = UNION_OFF;                           // 64*65536*4

    unsigned short* Tb  = (unsigned short*)(ws + TB_OFF);
    unsigned short* Xb  = (unsigned short*)(ws + XB_OFF);
    float*          al  = (float*)(ws + AL_OFF);
    float*          Z   = (float*)(ws + Z_OFF);
    float*          rl  = (float*)(ws + RL_OFF);
    float*          emt = (float*)(ws + EMT_OFF);
    unsigned short* Cb  = (unsigned short*)(ws + CB_OFF);
    unsigned short* Vb  = (unsigned short*)(ws + VB_OFF);
    float*          sc  = (float*)(ws + SC_OFF);
    float*          S   = (float*)(ws + S_OFF);

    // --- conversions ---
    conv_trans<<<(unsigned)(((long)R * EP + 255) / 256), 256, 0, stream>>>(tw, Tb);
    conv_x<<<(unsigned)(((long)(TT - 1) * NB * EP + 255) / 256), 256, 0, stream>>>(w, emb, Xb);
    conv_bf<<<(unsigned)(((long)KK * KK + 255) / 256), 256, 0, stream>>>(ecw, Cb, (long)KK * KK);
    conv_bf<<<(unsigned)(((long)VV * KK + 255) / 256), 256, 0, stream>>>(voc, Vb, (long)VV * KK);

    // --- emission table ---
    em_gemm<<<8 * (VV / 64), 256, 0, stream>>>(Cb, Vb, sc);
    row_lse_V<<<KK, 256, 0, stream>>>(sc, Z);
    em_gather<<<(unsigned)(((long)(TT - 1) * NB * KK + 255) / 256), 256, 0, stream>>>(sc, Z, w, emt);

    // --- initial alpha ---
    init_alpha<<<1, KK, 0, stream>>>(sw, sb, al);

    // --- forward scan ---
    for (int t = 1; t < TT; ++t) {
        const unsigned short* Xt = Xb + (size_t)(t - 1) * NB * EP;
        const float*          et = emt + (size_t)(t - 1) * NB * KK;
        step_gemm<<<R / (32 * CPB), 256, 0, stream>>>(Xt, Tb, S);
        row_logits<<<(NB * KK) / 8, 256, 0, stream>>>(S, al, rl);
        col_combine<<<NB, KK, 0, stream>>>(S, rl, et, al);
    }

    // --- final loss ---
    final_reduce<<<1, 256, 0, stream>>>(al, out);
    (void)in_sizes; (void)n_in; (void)out_size; (void)ws_size;
}
